// DualPointEncoder_51702816309282
// MI455X (gfx1250) — matrix operand order repaired and checked by reading, not compiled, not run on hardware
//
#include <hip/hip_runtime.h>

#define N_PTS   800000
#define BATCHES 8
#define P_PER   (N_PTS / BATCHES)   // 100000
#define K_SEL   4096
#define HIDDEN  32
#define CAP     8192                // candidate buffer per batch (>= K_SEL)
#define NB_RED  512                 // fixed reduction grid (deterministic)
#define EPSF    1e-5f

typedef _Float16 v16h __attribute__((ext_vector_type(16)));
typedef float    v8f  __attribute__((ext_vector_type(8)));

__device__ __forceinline__ v16h zero_v16h() {
  v16h v;
#pragma unroll
  for (int i = 0; i < 16; ++i) v[i] = (_Float16)0.0f;
  return v;
}
__device__ __forceinline__ v8f zero_v8f() {
  v8f v;
#pragma unroll
  for (int i = 0; i < 8; ++i) v[i] = 0.0f;
  return v;
}

// order-preserving float -> uint key (larger float => larger key)
__device__ __forceinline__ unsigned f2key(float f) {
  unsigned u = __float_as_uint(f);
  return (u & 0x80000000u) ? ~u : (u | 0x80000000u);
}

// ---------------------------------------------------------------- init scratch
__global__ void k_init(unsigned* hist, unsigned* prefix, unsigned* remaining, unsigned* cnt) {
  int b = blockIdx.x;
  hist[b * 256 + threadIdx.x] = 0u;
  if (threadIdx.x == 0) {
    prefix[b] = 0u;
    remaining[b] = (unsigned)K_SEL;
    cnt[b] = 0u;
  }
}

// ------------------------------------------------- stage 1: scoring + block max
__global__ void k_score(const float* __restrict__ sp, const float* __restrict__ inten,
                        const float* __restrict__ w1, const float* __restrict__ b1,
                        const float* __restrict__ bng, const float* __restrict__ bnb,
                        const float* __restrict__ bnm, const float* __restrict__ bnv,
                        const float* __restrict__ w2, const float* __restrict__ b2,
                        float* __restrict__ scores, float* __restrict__ pmax) {
  __shared__ float sm[256];
  float lmax = -3.4e38f;
  for (int i = blockIdx.x * blockDim.x + threadIdx.x; i < N_PTS;
       i += gridDim.x * blockDim.x) {
    if (i + 8192 < N_PTS) {  // emits global_prefetch_b8 on gfx1250
      __builtin_prefetch(&sp[(size_t)(i + 8192) * 3], 0, 1);
      __builtin_prefetch(&inten[i + 8192], 0, 1);
    }
    float c0 = sp[(size_t)i * 3 + 0], c1 = sp[(size_t)i * 3 + 1];
    float c2 = sp[(size_t)i * 3 + 2], c3 = inten[i];
    float s = b2[0];
#pragma unroll
    for (int j = 0; j < HIDDEN; ++j) {
      float z = c0 * w1[j] + c1 * w1[HIDDEN + j] + c2 * w1[2 * HIDDEN + j] +
                c3 * w1[3 * HIDDEN + j] + b1[j];
      z = (z - bnm[j]) * rsqrtf(bnv[j] + EPSF) * bng[j] + bnb[j];
      z = fmaxf(z, 0.0f);
      s += z * w2[j];
    }
    scores[i] = s;
    lmax = fmaxf(lmax, s);
  }
  sm[threadIdx.x] = lmax;
  __syncthreads();
  for (int o = 128; o > 0; o >>= 1) {
    if ((int)threadIdx.x < o) sm[threadIdx.x] = fmaxf(sm[threadIdx.x], sm[threadIdx.x + o]);
    __syncthreads();
  }
  if (threadIdx.x == 0) pmax[blockIdx.x] = sm[0];
}

__global__ void k_rmax(const float* __restrict__ pmax, float* __restrict__ gstat) {
  __shared__ float sm[NB_RED];
  sm[threadIdx.x] = pmax[threadIdx.x];
  __syncthreads();
  for (int o = NB_RED / 2; o > 0; o >>= 1) {
    if ((int)threadIdx.x < o) sm[threadIdx.x] = fmaxf(sm[threadIdx.x], sm[threadIdx.x + o]);
    __syncthreads();
  }
  if (threadIdx.x == 0) gstat[0] = sm[0];
}

__global__ void k_sumexp(const float* __restrict__ scores, const float* __restrict__ gstat,
                         float* __restrict__ psum) {
  __shared__ float sm[256];
  float gm = gstat[0];
  float acc = 0.0f;
  for (int i = blockIdx.x * blockDim.x + threadIdx.x; i < N_PTS;
       i += gridDim.x * blockDim.x)
    acc += __expf((scores[i] - gm) * 0.5f);
  sm[threadIdx.x] = acc;
  __syncthreads();
  for (int o = 128; o > 0; o >>= 1) {
    if ((int)threadIdx.x < o) sm[threadIdx.x] += sm[threadIdx.x + o];
    __syncthreads();
  }
  if (threadIdx.x == 0) psum[blockIdx.x] = sm[0];
}

__global__ void k_rsum(const float* __restrict__ psum, float* __restrict__ gstat) {
  __shared__ float sm[NB_RED];
  sm[threadIdx.x] = psum[threadIdx.x];
  __syncthreads();
  for (int o = NB_RED / 2; o > 0; o >>= 1) {
    if ((int)threadIdx.x < o) sm[threadIdx.x] += sm[threadIdx.x + o];
    __syncthreads();
  }
  if (threadIdx.x == 0) gstat[1] = sm[0];
}

// ------------------------------------- stage 2: per-batch radix-select (4 passes)
__global__ void k_hist(const float* __restrict__ scores, const unsigned* __restrict__ prefix,
                       unsigned* __restrict__ hist, int shift) {
  __shared__ unsigned lh[256];
  int b = blockIdx.y;
  lh[threadIdx.x] = 0u;
  __syncthreads();
  unsigned pre = prefix[b];
  unsigned hm = (shift == 24) ? 0u : (0xFFFFFFFFu << (shift + 8));
  const float* s = scores + (size_t)b * P_PER;
  for (int i = blockIdx.x * blockDim.x + threadIdx.x; i < P_PER;
       i += gridDim.x * blockDim.x) {
    unsigned key = f2key(s[i]);
    if ((key & hm) == pre) atomicAdd(&lh[(key >> shift) & 255u], 1u);
  }
  __syncthreads();
  if (lh[threadIdx.x]) atomicAdd(&hist[b * 256 + threadIdx.x], lh[threadIdx.x]);
}

__global__ void k_scan(unsigned* __restrict__ hist, unsigned* __restrict__ prefix,
                       unsigned* __restrict__ remaining, int shift) {
  __shared__ unsigned lh[256];
  int b = blockIdx.x;
  lh[threadIdx.x] = hist[b * 256 + threadIdx.x];
  __syncthreads();
  if (threadIdx.x == 0) {
    unsigned rem = remaining[b];
    unsigned cum = 0u;
    int d = 0;
    for (int bin = 255; bin >= 0; --bin) {
      unsigned c = lh[bin];
      if (cum + c >= rem) { d = bin; break; }
      cum += c;
    }
    remaining[b] = rem - cum;
    prefix[b] |= ((unsigned)d) << shift;
  }
  __syncthreads();
  hist[b * 256 + threadIdx.x] = 0u;  // clear for next pass
}

__global__ void k_compact(const float* __restrict__ scores, const unsigned* __restrict__ prefix,
                          unsigned* __restrict__ cnt, unsigned* __restrict__ candKey,
                          unsigned* __restrict__ candIdx) {
  int b = blockIdx.y;
  unsigned T = prefix[b];  // exact key of the K-th largest element
  const float* s = scores + (size_t)b * P_PER;
  for (int i = blockIdx.x * blockDim.x + threadIdx.x; i < P_PER;
       i += gridDim.x * blockDim.x) {
    unsigned key = f2key(s[i]);
    if (key >= T) {
      unsigned pos = atomicAdd(&cnt[b], 1u);
      if (pos < CAP) {
        candKey[b * CAP + pos] = key;
        candIdx[b * CAP + pos] = (unsigned)i;
      }
    }
  }
}

// sort candidates (key desc, index asc) in LDS; emit top-K indices per batch
__global__ __launch_bounds__(1024) void k_sort(const unsigned* __restrict__ candKey,
                                               const unsigned* __restrict__ candIdx,
                                               const unsigned* __restrict__ cnt,
                                               unsigned* __restrict__ selIdx) {
  __shared__ unsigned long long comp[CAP];  // 64KB (WGP has 320KB LDS)
  int b = blockIdx.x;
  unsigned m = cnt[b];
  if (m > CAP) m = CAP;
  for (int i = threadIdx.x; i < CAP; i += blockDim.x) {
    unsigned long long v = 0ull;
    if (i < (int)m)
      v = (((unsigned long long)candKey[b * CAP + i]) << 32) |
          (unsigned long long)(~candIdx[b * CAP + i]);
    comp[i] = v;
  }
  __syncthreads();
  for (int size = 2; size <= CAP; size <<= 1) {
    for (int stride = size >> 1; stride > 0; stride >>= 1) {
      for (int i = threadIdx.x; i < CAP / 2; i += blockDim.x) {
        int a = ((i / stride) * (stride << 1)) + (i & (stride - 1));
        int c = a + stride;
        bool descBlock = ((a & size) == 0);
        unsigned long long x = comp[a], y = comp[c];
        if ((x < y) == descBlock) { comp[a] = y; comp[c] = x; }
      }
      __syncthreads();
    }
  }
  for (int r = threadIdx.x; r < K_SEL; r += blockDim.x)
    selIdx[b * K_SEL + r] = ~((unsigned)(comp[r] & 0xFFFFFFFFull));
}

// sum of top-K (unnormalized) softmax numerators per batch
__global__ void k_wsum(const float* __restrict__ scores, const unsigned* __restrict__ selIdx,
                       const float* __restrict__ gstat, float* __restrict__ wsum) {
  __shared__ float sm[256];
  int b = blockIdx.x;
  float gm = gstat[0];
  float acc = 0.0f;
  for (int r = threadIdx.x; r < K_SEL; r += blockDim.x) {
    unsigned li = selIdx[b * K_SEL + r];
    acc += __expf((scores[(size_t)b * P_PER + li] - gm) * 0.5f);
  }
  sm[threadIdx.x] = acc;
  __syncthreads();
  for (int o = 128; o > 0; o >>= 1) {
    if ((int)threadIdx.x < o) sm[threadIdx.x] += sm[threadIdx.x + o];
    __syncthreads();
  }
  if (threadIdx.x == 0) wsum[b] = sm[0];
}

// gather selected features, scale by renormalized weight.
// norm_w = e / (sum_topk e + 1e-8 * Z)   (softmax denominator mostly cancels)
__global__ void k_gather(const float* __restrict__ sp, const float* __restrict__ inten,
                         const float* __restrict__ scores, const unsigned* __restrict__ selIdx,
                         const float* __restrict__ gstat, const float* __restrict__ wsum,
                         float* __restrict__ selbuf) {
  int row = blockIdx.x * blockDim.x + threadIdx.x;
  if (row >= BATCHES * K_SEL) return;
  int b = row / K_SEL;
  unsigned li = selIdx[row];
  size_t gi = (size_t)b * P_PER + li;
  float e = __expf((scores[gi] - gstat[0]) * 0.5f);
  float nw = e / (wsum[b] + 1e-8f * gstat[1]);
  selbuf[(size_t)row * 4 + 0] = sp[gi * 3 + 0] * nw;
  selbuf[(size_t)row * 4 + 1] = sp[gi * 3 + 1] * nw;
  selbuf[(size_t)row * 4 + 2] = sp[gi * 3 + 2] * nw;
  selbuf[(size_t)row * 4 + 3] = inten[gi] * nw;
}

// ------------------------------------- stage 3: fusion MLP via v_wmma_f32_16x16x32_f16
// One wave per 16-row tile. A layout per ISA 7.12.2 (16-bit 16x32):
//   lane l: M = l&15, half h = l>>4, VGPR v holds K = (v&3)*2 + 8h + 16*(v>=4), K+1
// B layout (32x16): lane l: N = l&15, VGPR v holds K = (v&3)*2 + 8h + 16*(v>=4), K+1 (same as A)
// C layout: lane l: N = l&15, VGPR j holds M = j + 8h
__global__ __launch_bounds__(256) void k_mlp(
    const float* __restrict__ selbuf, const float* __restrict__ fw1,
    const float* __restrict__ fb1, const float* __restrict__ g1, const float* __restrict__ bb1,
    const float* __restrict__ fw2, const float* __restrict__ fb2, const float* __restrict__ g2,
    const float* __restrict__ bb2, const float* __restrict__ fw3, const float* __restrict__ fb3,
    float* __restrict__ out) {
  __shared__ _Float16 st1[8 * 16 * 64];
  __shared__ _Float16 st2[8 * 16 * 32];
  const int lane = threadIdx.x & 31;
  const int wv = threadIdx.x >> 5;
  const int n = lane & 15;
  const int h = lane >> 4;
  const int rowBase = (blockIdx.x * 8 + wv) * 16;
  _Float16* a1 = st1 + wv * (16 * 64);
  _Float16* a2 = st2 + wv * (16 * 32);

  // ---- layer 1: [16x4] @ [4x64] (K padded 4 -> 32) ----
  v16h af = zero_v16h();
  if (h == 0) {  // K=0..3 live in v0/v1 of half 0 only
    const float* xr = selbuf + (size_t)(rowBase + n) * 4;
    af[0] = (_Float16)xr[0]; af[1] = (_Float16)xr[1];
    af[2] = (_Float16)xr[2]; af[3] = (_Float16)xr[3];
  }
  v8f c1[4];
#pragma unroll
  for (int t = 0; t < 4; ++t) {
    v16h bf = zero_v16h();
    if (h == 0) {
      bf[0] = (_Float16)fw1[0 * 64 + t * 16 + n];
      bf[1] = (_Float16)fw1[1 * 64 + t * 16 + n];
      bf[2] = (_Float16)fw1[2 * 64 + t * 16 + n];
      bf[3] = (_Float16)fw1[3 * 64 + t * 16 + n];
    }
    c1[t] = __builtin_amdgcn_wmma_f32_16x16x32_f16(false, af, false, bf, (short)0,
                                                   zero_v8f(), false, false);
  }
  // layernorm over 64 features, relu, stage to LDS as f16
  {
    float xs[4][8], psm[8], psq[8];
#pragma unroll
    for (int j = 0; j < 8; ++j) { psm[j] = 0.0f; psq[j] = 0.0f; }
#pragma unroll
    for (int t = 0; t < 4; ++t) {
      float bias = fb1[t * 16 + n];
#pragma unroll
      for (int j = 0; j < 8; ++j) {
        float x = c1[t][j] + bias;
        xs[t][j] = x; psm[j] += x; psq[j] += x * x;
      }
    }
#pragma unroll
    for (int j = 0; j < 8; ++j) {
#pragma unroll
      for (int msk = 1; msk < 16; msk <<= 1) {  // reduce across the 16-lane half
        psm[j] += __shfl_xor(psm[j], msk, 32);
        psq[j] += __shfl_xor(psq[j], msk, 32);
      }
    }
#pragma unroll
    for (int j = 0; j < 8; ++j) {
      float mean = psm[j] * (1.0f / 64.0f);
      float var = psq[j] * (1.0f / 64.0f) - mean * mean;
      float rstd = rsqrtf(var + EPSF);
      int m = j + 8 * h;
#pragma unroll
      for (int t = 0; t < 4; ++t) {
        int col = t * 16 + n;
        float a = fmaxf((xs[t][j] - mean) * rstd * g1[col] + bb1[col], 0.0f);
        a1[m * 64 + col] = (_Float16)a;
      }
    }
  }
  __syncthreads();

  // ---- layer 2: [16x64] @ [64x32] ----
  v16h a2f[2];
#pragma unroll
  for (int kk = 0; kk < 2; ++kk) {
#pragma unroll
    for (int v = 0; v < 8; ++v) {
      int Kb = kk * 32 + ((v & 3) * 2 + h * 8 + ((v >> 2) * 16));
      a2f[kk][2 * v]     = a1[n * 64 + Kb];
      a2f[kk][2 * v + 1] = a1[n * 64 + Kb + 1];
    }
  }
  v8f c2[2];
#pragma unroll
  for (int t = 0; t < 2; ++t) {
    c2[t] = zero_v8f();
#pragma unroll
    for (int kk = 0; kk < 2; ++kk) {
      v16h bf;
#pragma unroll
      for (int v = 0; v < 8; ++v) {
        int Kb = kk * 32 + ((v & 3) * 2 + h * 8 + ((v >> 2) * 16));
        bf[2 * v]     = (_Float16)fw2[Kb * 32 + t * 16 + n];
        bf[2 * v + 1] = (_Float16)fw2[(Kb + 1) * 32 + t * 16 + n];
      }
      c2[t] = __builtin_amdgcn_wmma_f32_16x16x32_f16(false, a2f[kk], false, bf, (short)0,
                                                     c2[t], false, false);
    }
  }
  // layernorm over 32 features, relu, stage to LDS
  {
    float xs[2][8], psm[8], psq[8];
#pragma unroll
    for (int j = 0; j < 8; ++j) { psm[j] = 0.0f; psq[j] = 0.0f; }
#pragma unroll
    for (int t = 0; t < 2; ++t) {
      float bias = fb2[t * 16 + n];
#pragma unroll
      for (int j = 0; j < 8; ++j) {
        float x = c2[t][j] + bias;
        xs[t][j] = x; psm[j] += x; psq[j] += x * x;
      }
    }
#pragma unroll
    for (int j = 0; j < 8; ++j) {
#pragma unroll
      for (int msk = 1; msk < 16; msk <<= 1) {
        psm[j] += __shfl_xor(psm[j], msk, 32);
        psq[j] += __shfl_xor(psq[j], msk, 32);
      }
    }
#pragma unroll
    for (int j = 0; j < 8; ++j) {
      float mean = psm[j] * (1.0f / 32.0f);
      float var = psq[j] * (1.0f / 32.0f) - mean * mean;
      float rstd = rsqrtf(var + EPSF);
      int m = j + 8 * h;
#pragma unroll
      for (int t = 0; t < 2; ++t) {
        int col = t * 16 + n;
        float a = fmaxf((xs[t][j] - mean) * rstd * g2[col] + bb2[col], 0.0f);
        a2[m * 32 + col] = (_Float16)a;
      }
    }
  }
  __syncthreads();

  // ---- layer 3: [16x32] @ [32x4] (N padded 4 -> 16) ----
  v16h a3f;
#pragma unroll
  for (int v = 0; v < 8; ++v) {
    int Kb = (v & 3) * 2 + h * 8 + ((v >> 2) * 16);
    a3f[2 * v]     = a2[n * 32 + Kb];
    a3f[2 * v + 1] = a2[n * 32 + Kb + 1];
  }
  v16h b3f = zero_v16h();
  if (n < 4) {
#pragma unroll
    for (int v = 0; v < 8; ++v) {
      int Kb = (v & 3) * 2 + h * 8 + ((v >> 2) * 16);
      b3f[2 * v]     = (_Float16)fw3[Kb * 4 + n];
      b3f[2 * v + 1] = (_Float16)fw3[(Kb + 1) * 4 + n];
    }
  }
  v8f c3 = __builtin_amdgcn_wmma_f32_16x16x32_f16(false, a3f, false, b3f, (short)0,
                                                  zero_v8f(), false, false);
  if (n < 4) {
    float bias = fb3[n];
#pragma unroll
    for (int j = 0; j < 8; ++j) {
      int m = j + 8 * h;
      out[(size_t)(rowBase + m) * 4 + n] = c3[j] + bias;
    }
  }
}

// ---------------------------------------------------------------------- launch
extern "C" void kernel_launch(void* const* d_in, const int* in_sizes, int n_in,
                              void* d_out, int out_size, void* d_ws, size_t ws_size,
                              hipStream_t stream) {
  const float* sp    = (const float*)d_in[0];
  const float* inten = (const float*)d_in[1];
  // d_in[2] = batch_idx (int64): sorted equal counts -> b = i / P, unused
  const float* w1  = (const float*)d_in[3];
  const float* b1  = (const float*)d_in[4];
  const float* bng = (const float*)d_in[5];
  const float* bnb = (const float*)d_in[6];
  const float* bnm = (const float*)d_in[7];
  const float* bnv = (const float*)d_in[8];
  const float* w2  = (const float*)d_in[9];
  const float* b2  = (const float*)d_in[10];
  const float* fw1 = (const float*)d_in[11];
  const float* fb1 = (const float*)d_in[12];
  const float* g1  = (const float*)d_in[13];
  const float* bb1 = (const float*)d_in[14];
  const float* fw2 = (const float*)d_in[15];
  const float* fb2 = (const float*)d_in[16];
  const float* g2  = (const float*)d_in[17];
  const float* bb2 = (const float*)d_in[18];
  const float* fw3 = (const float*)d_in[19];
  const float* fb3 = (const float*)d_in[20];
  float* out = (float*)d_out;

  // workspace carve-up (all 4-byte elements)
  float* wsf = (float*)d_ws;
  unsigned* wsu = (unsigned*)d_ws;
  size_t o = 0;
  float* scores       = wsf + o; o += N_PTS;
  float* pmax         = wsf + o; o += NB_RED;
  float* psum         = wsf + o; o += NB_RED;
  float* gstat        = wsf + o; o += 8;        // [0]=global max, [1]=Z
  unsigned* hist      = wsu + o; o += BATCHES * 256;
  unsigned* prefix    = wsu + o; o += BATCHES;
  unsigned* remaining = wsu + o; o += BATCHES;
  unsigned* cnt       = wsu + o; o += BATCHES;
  unsigned* candKey   = wsu + o; o += BATCHES * CAP;
  unsigned* candIdx   = wsu + o; o += BATCHES * CAP;
  unsigned* selIdx    = wsu + o; o += BATCHES * K_SEL;
  float* wsumTop      = wsf + o; o += BATCHES;
  float* selbuf       = wsf + o; o += (size_t)BATCHES * K_SEL * 4;

  k_init<<<BATCHES, 256, 0, stream>>>(hist, prefix, remaining, cnt);
  k_score<<<NB_RED, 256, 0, stream>>>(sp, inten, w1, b1, bng, bnb, bnm, bnv, w2, b2,
                                      scores, pmax);
  k_rmax<<<1, NB_RED, 0, stream>>>(pmax, gstat);
  k_sumexp<<<NB_RED, 256, 0, stream>>>(scores, gstat, psum);
  k_rsum<<<1, NB_RED, 0, stream>>>(psum, gstat);

  const int shifts[4] = {24, 16, 8, 0};
  for (int p = 0; p < 4; ++p) {
    k_hist<<<dim3(64, BATCHES), 256, 0, stream>>>(scores, prefix, hist, shifts[p]);
    k_scan<<<BATCHES, 256, 0, stream>>>(hist, prefix, remaining, shifts[p]);
  }
  k_compact<<<dim3(64, BATCHES), 256, 0, stream>>>(scores, prefix, cnt, candKey, candIdx);
  k_sort<<<BATCHES, 1024, 0, stream>>>(candKey, candIdx, cnt, selIdx);
  k_wsum<<<BATCHES, 256, 0, stream>>>(scores, selIdx, gstat, wsumTop);
  k_gather<<<(BATCHES * K_SEL + 255) / 256, 256, 0, stream>>>(sp, inten, scores, selIdx,
                                                              gstat, wsumTop, selbuf);
  k_mlp<<<(BATCHES * K_SEL) / 128, 256, 0, stream>>>(selbuf, fw1, fb1, g1, bb1, fw2, fb2,
                                                     g2, bb2, fw3, fb3, out);
}
